// TabularMambaRegressor_38225208934427
// MI455X (gfx1250) — compile-verified, matrix operand order checked by reading, not run on hardware
//
#include <hip/hip_runtime.h>

typedef unsigned short u16;
typedef __attribute__((ext_vector_type(16))) __bf16 v16bf;
typedef __attribute__((ext_vector_type(8)))  float  v8f;

constexpr int K_B   = 256;   // batch
constexpr int K_F   = 255;   // numeric features
constexpr int K_L   = 256;   // sequence length (F+1)
constexpr int K_DM  = 256;   // d_model
constexpr int K_DI  = 512;   // d_inner
constexpr int K_DS  = 16;    // d_state
constexpr int K_DTR = 16;    // dt_rank
constexpr int K_NL  = 4;     // layers
constexpr int K_T   = K_B * K_L;  // 65536 tokens
constexpr int K_XP  = 64;    // padded x_proj width (dt_rank + 2*d_state = 48 -> 64)

// ---------- bf16 helpers (storage = u16) ----------
__device__ __forceinline__ float bf2f(u16 u) {
  union { unsigned int i; float f; } c; c.i = ((unsigned int)u) << 16; return c.f;
}
__device__ __forceinline__ u16 f2bf(float f) {
  union { float f; unsigned int i; } c; c.f = f;
  unsigned int r = c.i + 0x7FFFu + ((c.i >> 16) & 1u);
  return (u16)(r >> 16);
}

// ---------- WMMA 16x32 bf16 operand loader (no guards: shapes are padded) ----
// ISA 16-bit operand layout (16x32): lane L holds row L%16, half h=L/16.
// elements 0..7  -> K = h*8 + e        (one contiguous 16B load)
// elements 8..15 -> K = 16 + h*8 + e-8 (one contiguous 16B load)
union TileHalf { uint4 q[2]; v16bf v; };

__device__ __forceinline__ v16bf load_tile(const u16* __restrict__ p) {
  TileHalf t;
  t.q[0] = *(const uint4*)(p);
  t.q[1] = *(const uint4*)(p + 16);
  return t.v;
}

__device__ __forceinline__ v8f wmma_bf16(v16bf a, v16bf b, v8f c) {
  return __builtin_amdgcn_wmma_f32_16x16x32_bf16(false, a, false, b, (short)0, c,
                                                 false, false);
}

// ---------- WMMA GEMM: C[M,N] = A[M,K] * W[N,K]^T ----------
// A, W bf16 row-major. Requires M % 128 == 0, N % 64 == 0, K % 64 == 0.
// grid = (N/64, M/128); block = 256 threads = 8 waves.
// Wave w computes a 16(M) x 64(N) strip: 4 accumulators, A tile reused 4x.
// Explicit double-buffered pipeline: tiles for step k+32 are in flight while
// the 4 WMMAs for step k execute.
__global__ void gemm_wmma(const u16* __restrict__ A, int lda,
                          const u16* __restrict__ W, int ldw,
                          float* __restrict__ Cf, u16* __restrict__ Cb, int ldc,
                          int K) {
  const int lane = threadIdx.x & 31;
  const int wave = threadIdx.x >> 5;
  const long m0 = ((long)blockIdx.y * 8 + wave) * 16;
  const long n0 = (long)blockIdx.x * 64;
  const int r = lane & 15;
  const int h = lane >> 4;

  const u16* ap = A + (size_t)(m0 + r) * (size_t)lda + h * 8;
  const u16* wp = W + (size_t)(n0 + r) * (size_t)ldw + h * 8;
  const size_t ws1 = (size_t)16 * (size_t)ldw;

  // one-shot warm-up prefetch of the strips (pipeline covers the rest)
  __builtin_prefetch(ap, 0, 3);
  __builtin_prefetch(wp, 0, 3);
  __builtin_prefetch(wp + 2 * ws1, 0, 3);

  v8f acc0 = {0.f,0.f,0.f,0.f,0.f,0.f,0.f,0.f};
  v8f acc1 = acc0, acc2 = acc0, acc3 = acc0;

  // prologue: load tiles for k = 0
  v16bf a  = load_tile(ap);
  v16bf b0 = load_tile(wp);
  v16bf b1 = load_tile(wp + ws1);
  v16bf b2 = load_tile(wp + 2 * ws1);
  v16bf b3 = load_tile(wp + 3 * ws1);

#pragma unroll 2
  for (int k0 = 32; k0 < K; k0 += 32) {
    // issue next-step loads first; they retire while the WMMAs below execute
    v16bf an  = load_tile(ap + k0);
    v16bf b0n = load_tile(wp + k0);
    v16bf b1n = load_tile(wp + ws1 + k0);
    v16bf b2n = load_tile(wp + 2 * ws1 + k0);
    v16bf b3n = load_tile(wp + 3 * ws1 + k0);
    acc0 = wmma_bf16(a, b0, acc0);
    acc1 = wmma_bf16(a, b1, acc1);
    acc2 = wmma_bf16(a, b2, acc2);
    acc3 = wmma_bf16(a, b3, acc3);
    a = an; b0 = b0n; b1 = b1n; b2 = b2n; b3 = b3n;
  }
  // epilogue
  acc0 = wmma_bf16(a, b0, acc0);
  acc1 = wmma_bf16(a, b1, acc1);
  acc2 = wmma_bf16(a, b2, acc2);
  acc3 = wmma_bf16(a, b3, acc3);

  // C/D layout: N = lane%16, M = vgpr + 8*(lane/16)
  const v8f accs[4] = {acc0, acc1, acc2, acc3};
#pragma unroll
  for (int j = 0; j < 4; ++j) {
    const long n = n0 + j * 16 + r;
#pragma unroll
    for (int v = 0; v < 8; ++v) {
      const long m = m0 + v + 8 * h;
      const float x = accs[j][v];
      if (Cf) Cf[(size_t)m * (size_t)ldc + n] = x;
      if (Cb) Cb[(size_t)m * (size_t)ldc + n] = f2bf(x);
    }
  }
}

// ---------- fp32 -> bf16 conversion ----------
__global__ void f32_to_bf16(const float* __restrict__ src, u16* __restrict__ dst, int n) {
  int i = blockIdx.x * blockDim.x + threadIdx.x;
  if (i < n) dst[i] = f2bf(src[i]);
}

// ---------- W_xproj: convert + zero-pad N 48 -> 64 ----------
// dst [NL][64][512], src [NL][48][512]
__global__ void pad_wx(const float* __restrict__ src, u16* __restrict__ dst) {
  const int nrow = blockIdx.x;         // 0 .. NL*64-1
  const int k = threadIdx.x;           // 0..511
  const int i = nrow >> 6;
  const int n = nrow & 63;
  dst[(size_t)nrow * K_DI + k] =
      (n < 48) ? f2bf(src[((size_t)i * 48 + n) * K_DI + k]) : (u16)0;
}

// ---------- build token matrix (embedding row + scalar projection) ----------
__global__ void build_x(const float* __restrict__ x_num, const int* __restrict__ loc_ids,
                        const float* __restrict__ W_scal, const float* __restrict__ b_scal,
                        const float* __restrict__ emb, u16* __restrict__ xb) {
  const int t = blockIdx.x;        // token = b*L + l
  const int d = threadIdx.x;       // 0..255
  const int b = t / K_L;
  const int l = t % K_L;
  float v;
  if (l == 0) {
    v = emb[(size_t)loc_ids[b] * K_DM + d];
  } else {
    v = x_num[(size_t)b * K_F + (l - 1)] * W_scal[d] + b_scal[d];
  }
  xb[(size_t)t * K_DM + d] = f2bf(v);
}

// ---------- causal depthwise conv (width 4) + SiLU ----------
__global__ void conv_silu(const u16* __restrict__ xz, const float* __restrict__ cw,
                          const float* __restrict__ cb, u16* __restrict__ u) {
  const int t = blockIdx.x;        // token = b*L + l
  const int e = threadIdx.x;       // channel 0..511
  const int l = t % K_L;
  float acc = cb[e];
#pragma unroll
  for (int j = 0; j < 4; ++j) {
    const int ll = l - 3 + j;
    if (ll >= 0) {
      const int tt = t - 3 + j;
      acc += bf2f(xz[(size_t)tt * (2 * K_DI) + e]) * cw[e * 4 + j];
    }
  }
  const float sig = 1.f / (1.f + __expf(-acc));
  u[(size_t)t * K_DI + e] = f2bf(acc * sig);
}

// ---------- selective scan (fused dt_proj + softplus + SSM + D-skip + gate) ----------
// one block per batch element; one thread per inner channel; h[16] state in VGPRs
__global__ void scan_kernel(const float* __restrict__ xdbl, const u16* __restrict__ u,
                            const u16* __restrict__ xz, const float* __restrict__ Wdt,
                            const float* __restrict__ bdt, const float* __restrict__ Alog,
                            const float* __restrict__ Dskip, u16* __restrict__ y) {
  const int b = blockIdx.x;
  const int e = threadIdx.x;       // 0..511
  __shared__ float sh[48];         // dt[16] | B[16] | C[16] for the current step

  float A_row[K_DS], Wdt_row[K_DTR], h[K_DS];
#pragma unroll
  for (int s = 0; s < K_DS; ++s) {
    A_row[s] = -__expf(Alog[(size_t)e * K_DS + s]);
    h[s] = 0.f;
  }
#pragma unroll
  for (int r = 0; r < K_DTR; ++r) Wdt_row[r] = Wdt[(size_t)e * K_DTR + r];
  const float bdt_e = bdt[e];
  const float D_e = Dskip[e];

  for (int l = 0; l < K_L; ++l) {
    const size_t t = (size_t)b * K_L + l;
    if (e < 48) sh[e] = xdbl[t * K_XP + e];
    __syncthreads();

    float dtv = bdt_e;
#pragma unroll
    for (int r = 0; r < K_DTR; ++r) dtv += sh[r] * Wdt_row[r];
    const float delta = (dtv > 20.f) ? dtv : log1pf(__expf(dtv));

    const float u_t = bf2f(u[t * K_DI + e]);
    const float du = delta * u_t;
    float yacc = 0.f;
#pragma unroll
    for (int s = 0; s < K_DS; ++s) {
      h[s] = __expf(delta * A_row[s]) * h[s] + du * sh[16 + s];
      yacc += h[s] * sh[32 + s];
    }

    const float z = bf2f(xz[t * (2 * K_DI) + K_DI + e]);
    const float gate = z / (1.f + __expf(-z));
    y[t * K_DI + e] = f2bf((yacc + u_t * D_e) * gate);
    __syncthreads();
  }
}

// ---------- LayerNorm over d + mean-pool over L (wave-per-row, shfl reduce) ----
// block = 256 threads = 8 waves; wave w handles rows l = w, w+8, ...
// lane owns 8 contiguous d's; mean/var via wave32 shuffles; pooled acc in regs.
__global__ void ln_pool(const float* __restrict__ x, const float* __restrict__ g,
                        const float* __restrict__ bta, u16* __restrict__ pooled) {
  const int b = blockIdx.x;
  const int lane = threadIdx.x & 31;
  const int wave = threadIdx.x >> 5;   // 0..7
  __shared__ float accs[8][K_DM];

  const int d0 = lane * 8;
  float gv[8], bv[8], acc[8];
#pragma unroll
  for (int j = 0; j < 8; ++j) {
    gv[j] = g[d0 + j];
    bv[j] = bta[d0 + j];
    acc[j] = 0.f;
  }

  for (int l = wave; l < K_L; l += 8) {
    const float* row = x + ((size_t)b * K_L + l) * K_DM + d0;
    float v[8];
#pragma unroll
    for (int j = 0; j < 8; ++j) v[j] = row[j];
    float s = 0.f;
#pragma unroll
    for (int j = 0; j < 8; ++j) s += v[j];
#pragma unroll
    for (int m = 16; m >= 1; m >>= 1) s += __shfl_xor(s, m, 32);
    const float mu = s * (1.f / K_DM);
    float s2 = 0.f;
#pragma unroll
    for (int j = 0; j < 8; ++j) { const float dv = v[j] - mu; s2 += dv * dv; }
#pragma unroll
    for (int m = 16; m >= 1; m >>= 1) s2 += __shfl_xor(s2, m, 32);
    const float rs = rsqrtf(s2 * (1.f / K_DM) + 1e-5f);
#pragma unroll
    for (int j = 0; j < 8; ++j) acc[j] += (v[j] - mu) * rs * gv[j] + bv[j];
  }

#pragma unroll
  for (int j = 0; j < 8; ++j) accs[wave][d0 + j] = acc[j];
  __syncthreads();

  const int d = threadIdx.x;
  float t = 0.f;
#pragma unroll
  for (int w = 0; w < 8; ++w) t += accs[w][d];
  pooled[(size_t)b * K_DM + d] = f2bf(t * (1.f / K_L));
}

// ---------- head: bias + exact GELU + final dot ----------
__global__ void head_out(const float* __restrict__ hbuf, const float* __restrict__ bh1,
                         const float* __restrict__ W2, const float* __restrict__ b2,
                         float* __restrict__ out) {
  const int b = blockIdx.x;
  const int d = threadIdx.x;       // 0..255
  __shared__ float red[K_DM];
  const float x = hbuf[(size_t)b * K_DM + d] + bh1[d];
  const float gelu = 0.5f * x * (1.f + erff(x * 0.70710678118654752f));
  red[d] = gelu * W2[d];
  __syncthreads();
  for (int s = 128; s > 0; s >>= 1) {
    if (d < s) red[d] += red[d + s];
    __syncthreads();
  }
  if (d == 0) out[b] = red[0] + b2[0];
}

// ---------------------------------------------------------------------------
extern "C" void kernel_launch(void* const* d_in, const int* in_sizes, int n_in,
                              void* d_out, int out_size, void* d_ws, size_t ws_size,
                              hipStream_t stream) {
  (void)in_sizes; (void)n_in; (void)out_size; (void)ws_size;
  const float* x_num   = (const float*)d_in[0];
  const int*   loc_ids = (const int*)d_in[1];
  const float* W_scal  = (const float*)d_in[2];
  const float* b_scal  = (const float*)d_in[3];
  const float* emb     = (const float*)d_in[4];
  const float* W_in    = (const float*)d_in[5];
  const float* conv_w  = (const float*)d_in[6];
  const float* conv_b  = (const float*)d_in[7];
  const float* W_xp    = (const float*)d_in[8];
  const float* W_dt    = (const float*)d_in[9];
  const float* b_dt    = (const float*)d_in[10];
  const float* A_log   = (const float*)d_in[11];
  const float* D_skip  = (const float*)d_in[12];
  const float* W_out   = (const float*)d_in[13];
  const float* ln_g    = (const float*)d_in[14];
  const float* ln_b    = (const float*)d_in[15];
  const float* W_h1    = (const float*)d_in[16];
  const float* b_h1    = (const float*)d_in[17];
  const float* W_h2    = (const float*)d_in[18];
  const float* b_h2    = (const float*)d_in[19];
  float* out = (float*)d_out;

  // workspace carving (all offsets 256B-aligned)
  char* ws = (char*)d_ws;
  size_t off = 0;
  auto carve = [&](size_t bytes) -> char* {
    char* p = ws + off;
    off += (bytes + 255) & ~(size_t)255;
    return p;
  };
  u16*   xb     = (u16*)carve((size_t)K_T * K_DM * 2);          // layer input, bf16
  u16*   xz     = (u16*)carve((size_t)K_T * 2 * K_DI * 2);      // in_proj out, bf16
  u16*   u      = (u16*)carve((size_t)K_T * K_DI * 2);          // conv+silu out, bf16
  float* xdbl   = (float*)carve((size_t)K_T * K_XP * 4);        // x_proj out (padded), fp32
  u16*   y      = (u16*)carve((size_t)K_T * K_DI * 2);          // scan out, bf16
  float* xoutf  = (float*)carve((size_t)K_T * K_DM * 4);        // last-layer out, fp32
  u16*   Win_b  = (u16*)carve((size_t)K_NL * 1024 * 256 * 2);
  u16*   Wx_b   = (u16*)carve((size_t)K_NL * K_XP * 512 * 2);   // padded to 64 rows
  u16*   Wo_b   = (u16*)carve((size_t)K_NL * 256 * 512 * 2);
  u16*   Wh1_b  = (u16*)carve((size_t)256 * 256 * 2);
  u16*   pooled = (u16*)carve((size_t)256 * 256 * 2);
  float* hbuf   = (float*)carve((size_t)256 * 256 * 4);

  // 1. weight conversion to bf16 (+ padding for x_proj)
  auto cvt = [&](const float* src, u16* dst, int n) {
    f32_to_bf16<<<(n + 255) / 256, 256, 0, stream>>>(src, dst, n);
  };
  cvt(W_in,  Win_b, K_NL * 1024 * 256);
  cvt(W_out, Wo_b,  K_NL * 256 * 512);
  cvt(W_h1,  Wh1_b, 256 * 256);
  pad_wx<<<K_NL * K_XP, K_DI, 0, stream>>>(W_xp, Wx_b);

  // 2. token matrix
  build_x<<<K_T, K_DM, 0, stream>>>(x_num, loc_ids, W_scal, b_scal, emb, xb);

  // 3. mamba layers
  for (int i = 0; i < K_NL; ++i) {
    {  // in_proj: (T,256) x (1024,256)^T -> xz bf16
      dim3 g((2 * K_DI) / 64, K_T / 128);
      gemm_wmma<<<g, 256, 0, stream>>>(xb, K_DM, Win_b + (size_t)i * 1024 * 256, K_DM,
                                       nullptr, xz, 2 * K_DI, K_DM);
    }
    conv_silu<<<K_T, K_DI, 0, stream>>>(xz, conv_w + (size_t)i * K_DI * 4,
                                        conv_b + (size_t)i * K_DI, u);
    {  // x_proj: (T,512) x (64,512)^T -> xdbl fp32 (padded)
      dim3 g(K_XP / 64, K_T / 128);
      gemm_wmma<<<g, 256, 0, stream>>>(u, K_DI, Wx_b + (size_t)i * K_XP * 512, K_DI,
                                       xdbl, nullptr, K_XP, K_DI);
    }
    scan_kernel<<<K_B, K_DI, 0, stream>>>(xdbl, u, xz,
                                          W_dt + (size_t)i * K_DI * K_DTR,
                                          b_dt + (size_t)i * K_DI,
                                          A_log + (size_t)i * K_DI * K_DS,
                                          D_skip + (size_t)i * K_DI, y);
    {  // out_proj: (T,512) x (256,512)^T
      dim3 g(K_DM / 64, K_T / 128);
      if (i < K_NL - 1) {
        gemm_wmma<<<g, 256, 0, stream>>>(y, K_DI, Wo_b + (size_t)i * 256 * 512, K_DI,
                                         nullptr, xb, K_DM, K_DI);
      } else {
        gemm_wmma<<<g, 256, 0, stream>>>(y, K_DI, Wo_b + (size_t)i * 256 * 512, K_DI,
                                         xoutf, nullptr, K_DM, K_DI);
      }
    }
  }

  // 4. layernorm + pool
  ln_pool<<<K_B, K_DM, 0, stream>>>(xoutf, ln_g, ln_b, pooled);

  // 5. head GEMM: (256,256) x (256,256)^T -> hbuf fp32
  {
    dim3 g(K_DM / 64, K_B / 128);
    gemm_wmma<<<g, 256, 0, stream>>>(pooled, K_DM, Wh1_b, K_DM,
                                     hbuf, nullptr, K_DM, K_DM);
  }

  // 6. bias + GELU + final dot
  head_out<<<K_B, K_DM, 0, stream>>>(hbuf, b_h1, W_h2, b_h2, out);
}
